// BertSelfAttention_6957847019918
// MI455X (gfx1250) — matrix operand order repaired and checked by reading, not compiled, not run on hardware
//
#include <hip/hip_runtime.h>
#include <hip/hip_bf16.h>

#ifndef __has_builtin
#define __has_builtin(x) 0
#endif

#if __has_builtin(__builtin_amdgcn_global_load_async_to_lds_b128) && \
    __has_builtin(__builtin_amdgcn_s_wait_asynccnt)
#define USE_ASYNC 1
#else
#define USE_ASYNC 0
#endif

typedef __attribute__((ext_vector_type(16))) __bf16 v16bf;
typedef __attribute__((ext_vector_type(8)))  float  v8f;
typedef __attribute__((ext_vector_type(8)))  unsigned int v8u;
typedef __attribute__((ext_vector_type(4)))  int    v4i;

typedef __attribute__((address_space(1))) v4i* gas_v4i_ptr;
typedef __attribute__((address_space(3))) v4i* las_v4i_ptr;

__device__ __forceinline__ unsigned short f2bfbits(float f) {
  unsigned int u = __float_as_uint(f);
  unsigned int r = u + 0x7FFFu + ((u >> 16) & 1u);   // round-to-nearest-even
  return (unsigned short)(r >> 16);
}

__device__ __forceinline__ unsigned int pk_bf16(float lo, float hi) {
  return (unsigned int)f2bfbits(lo) | ((unsigned int)f2bfbits(hi) << 16);
}

__device__ __forceinline__ v8f wmma_bf16(v8u a, v8u b, v8f c) {
  return __builtin_amdgcn_wmma_f32_16x16x32_bf16(
      false, __builtin_bit_cast(v16bf, a),
      false, __builtin_bit_cast(v16bf, b),
      (short)0, c, false, false);
}

// 16-byte global -> LDS copy: async DMA (ASYNCcnt) when available.
__device__ __forceinline__ void async_copy16(const void* g, void* l) {
#if USE_ASYNC
  __builtin_amdgcn_global_load_async_to_lds_b128(
      (gas_v4i_ptr)g, (las_v4i_ptr)l, 0, 0);
#else
  *(uint4*)l = *(const uint4*)g;
#endif
}
__device__ __forceinline__ void wait_async_all() {
#if USE_ASYNC
  __builtin_amdgcn_s_wait_asynccnt(0);
#endif
}

// ---------------------------------------------------------------------------
// Prep 0: X fp32 -> bf16, flat.  8 elems / thread.
// ---------------------------------------------------------------------------
__global__ __launch_bounds__(256) void prep_x(
    const float* __restrict__ X, unsigned short* __restrict__ Xbf)
{
  size_t base = ((size_t)blockIdx.x * 256 + threadIdx.x) * 8;
  unsigned int* o32 = (unsigned int*)Xbf;
#pragma unroll
  for (int i = 0; i < 8; i += 2)
    o32[(base + i) >> 1] = pk_bf16(X[base + i], X[base + i + 1]);
}

// ---------------------------------------------------------------------------
// Prep 1: W[k][n] fp32 -> Wt[z][n][k] bf16, 32x32 LDS-tiled transpose.
// grid (32, 32, 3), block 256.
// ---------------------------------------------------------------------------
__global__ __launch_bounds__(256) void prep_w(
    const float* __restrict__ Wq, const float* __restrict__ Wk,
    const float* __restrict__ Wv, unsigned short* __restrict__ Wt)
{
  __shared__ float t[32][33];
  const int z = blockIdx.z;
  const float* W = (z == 0) ? Wq : (z == 1) ? Wk : Wv;
  unsigned short* o = Wt + (size_t)z * 1024 * 1024;

  const int tx = threadIdx.x & 31, ty = threadIdx.x >> 5;   // ty in 0..7
  const int n0 = blockIdx.x * 32, k0 = blockIdx.y * 32;
#pragma unroll
  for (int r = 0; r < 4; ++r)                    // coalesced read along n
    t[ty + 8 * r][tx] = W[(size_t)(k0 + ty + 8 * r) * 1024 + n0 + tx];
  __syncthreads();
#pragma unroll
  for (int r = 0; r < 4; ++r)                    // coalesced write along k
    o[(size_t)(n0 + ty + 8 * r) * 1024 + k0 + tx] = f2bfbits(t[tx][ty + 8 * r]);
}

// ---------------------------------------------------------------------------
// Kernel 1: QKV projection GEMM, all-bf16 data path.
// Xbf[4096,1024] @ Wt[z][n][k] + b -> bf16.  z=0:Q, z=1:K row-major
// [B*S,1024]; z=2:V transposed [B,1024,2048].  Block = 256 thr (8 waves),
// block tile 128x64, wave tile 32x32.  Triple-buffered async LDS staging,
// one barrier per K-step; fragments are pure ds_load_b128, no conversions.
// ---------------------------------------------------------------------------
__global__ __launch_bounds__(256) void qkv_gemm(
    const unsigned short* __restrict__ Xbf,   // [4096,1024] bf16
    const unsigned short* __restrict__ Wt,    // [3][1024 n][1024 k] bf16
    const float* __restrict__ bq, const float* __restrict__ bk,
    const float* __restrict__ bv,
    unsigned short* __restrict__ qo, unsigned short* __restrict__ ko,
    unsigned short* __restrict__ vo)
{
  constexpr int K = 1024, N = 1024, SROW = 2048;
  __shared__ __align__(16) unsigned short sA[3][128 * 32];  // [m][k] 8KB x3
  __shared__ __align__(16) unsigned short sB[3][64 * 32];   // [n][k] 4KB x3

  const int z = blockIdx.z;
  const float* bias = (z == 0) ? bq : (z == 1) ? bk : bv;
  unsigned short* out = (z == 0) ? qo : (z == 1) ? ko : vo;
  const unsigned short* Wz = Wt + (size_t)z * N * K;

  const int tid  = threadIdx.x;
  const int lane = tid & 31, wid = tid >> 5;
  const int m0 = (wid >> 1) * 32;
  const int n0 = (wid & 1) * 32;
  const int blockM = blockIdx.x * 128;
  const int blockN = blockIdx.y * 64;
  const int fm = lane & 15, hi = lane >> 4;
  const int kb = hi * 8, kb2 = hi * 16;

  auto stage = [&](int buf, int kt) {
#pragma unroll
    for (int i = 0; i < 2; ++i) {                // X tile: 512 x 16B chunks
      int idx = i * 256 + tid;
      int m = idx >> 2, ch = idx & 3;
      async_copy16(&Xbf[(size_t)(blockM + m) * K + kt + ch * 8],
                   &sA[buf][idx * 8]);
    }
    {                                            // Wt tile: 256 x 16B chunks
      int n = tid >> 2, ch = tid & 3;
      async_copy16(&Wz[(size_t)(blockN + n) * K + kt + ch * 8],
                   &sB[buf][n * 32 + ch * 8]);
    }
  };

  v8f acc[2][2] = {};
  stage(0, 0);

  for (int kt = 0; kt < K; kt += 32) {
    const int cur = (kt >> 5) % 3;
    wait_async_all();
    __syncthreads();
    if (kt + 32 < K) stage((cur + 1) % 3, kt + 32);

    const unsigned int* lA32 = (const unsigned int*)sA[cur];
    const unsigned int* lB32 = (const unsigned int*)sB[cur];

    v8u afr[2], bfr[2];
#pragma unroll
    for (int i = 0; i < 2; ++i)
#pragma unroll
      for (int v = 0; v < 8; ++v) {
        int k = kb2 + 2 * v;
        afr[i][v] = lA32[((m0 + 16 * i + fm) * 32 + k) >> 1];
      }
#pragma unroll
    for (int j = 0; j < 2; ++j)
#pragma unroll
      for (int v = 0; v < 8; ++v)
        bfr[j][v] = lB32[((n0 + 16 * j + fm) * 32 + kb2 + 2 * v) >> 1];

#pragma unroll
    for (int i = 0; i < 2; ++i)
#pragma unroll
      for (int j = 0; j < 2; ++j)
        acc[i][j] = wmma_bf16(afr[i], bfr[j], acc[i][j]);
  }

#pragma unroll
  for (int j = 0; j < 2; ++j) {
    int col = blockN + n0 + 16 * j + fm;
    float bv_ = bias[col];
#pragma unroll
    for (int i = 0; i < 2; ++i)
#pragma unroll
      for (int v = 0; v < 8; ++v) {
        int mg = blockM + m0 + 16 * i + hi * 8 + v;
        unsigned short bits = f2bfbits(acc[i][j][v] + bv_);
        if (z < 2) {
          out[(size_t)mg * N + col] = bits;
        } else {                                  // V transposed [B,1024,2048]
          int bb = mg >> 11, s = mg & (SROW - 1);
          out[((size_t)(bb * N + col)) * SROW + s] = bits;
        }
      }
  }
}

// ---------------------------------------------------------------------------
// Kernel 2: fused flash attention.  8 waves/block share one (b,h); K/V 32-key
// tiles are async-DMA'd into triple-buffered LDS (one barrier per iteration),
// each wave runs a 16-query online-softmax tile.  grid = (S/128, B*H).
// ---------------------------------------------------------------------------
__global__ __launch_bounds__(256) void attn(
    const unsigned short* __restrict__ Q,   // [B*S,1024] bf16
    const unsigned short* __restrict__ Km,  // [B*S,1024] bf16
    const unsigned short* __restrict__ Vt,  // [B,1024,2048] bf16 (transposed)
    const float* __restrict__ mask,         // [B,S]
    float* __restrict__ out)                // [B*S,1024] f32
{
  constexpr int S = 2048, H = 1024, HD = 64;
  __shared__ __align__(16) unsigned short lK[3][32 * 64];  // [key][d] 4KB x3
  __shared__ __align__(16) unsigned short lV[3][64 * 32];  // [d][key] 4KB x3
  __shared__ unsigned int lP[8][16 * 16];                  // per-wave P buffer

  const int tid = threadIdx.x, lane = tid & 31, wid = tid >> 5;
  const int bh = blockIdx.y, b = bh >> 4, h = bh & 15;
  const int q0 = blockIdx.x * 128 + wid * 16;
  const int fm = lane & 15, hi = lane >> 4;
  const int kb = hi * 8;

  const unsigned short* qbase = Q  + (size_t)(b * S) * H + h * HD;
  const unsigned short* kbase = Km + (size_t)(b * S) * H + h * HD;
  const unsigned short* vbase = Vt + ((size_t)b * H + h * HD) * S;
  const float* mbase = mask + b * S;

  // Q A-fragments for d=[0,32) and d=[32,64)
  v8u qa[2];
#pragma unroll
  for (int c = 0; c < 2; ++c)
#pragma unroll
    for (int v = 0; v < 8; ++v) {
      int d = 32 * c + hi * 16 + 2 * v;
      qa[c][v] = *(const unsigned int*)(qbase + (size_t)(q0 + fm) * H + d);
    }

  auto stageKV = [&](int buf, int kk) {
    {  // K tile: 32 keys x 64 d (row-major), 256 x 16B
      int key = tid >> 3, ch = tid & 7;
      async_copy16(kbase + (size_t)(kk + key) * H + ch * 8,
                   &lK[buf][key * 64 + ch * 8]);
    }
    {  // V tile: 64 d x 32 keys (transposed layout), 256 x 16B
      int d = tid >> 2, ch = tid & 3;
      async_copy16(vbase + (size_t)d * S + kk + ch * 8,
                   &lV[buf][d * 32 + ch * 8]);
    }
  };

  float mrow[8], lrow[8];
  v8f accO[4] = {};
#pragma unroll
  for (int v = 0; v < 8; ++v) { mrow[v] = -3.0e38f; lrow[v] = 0.0f; }

  stageKV(0, 0);
  for (int kk = 0; kk < S; kk += 32) {
    const int cur = (kk >> 5) % 3;
    wait_async_all();          // our part of tile(cur) is in LDS
    __syncthreads();           // everyone's part is in LDS
    if (kk + 32 < S) stageKV((cur + 1) % 3, kk + 32);  // overlap next DMA

    const unsigned int* lK32 = (const unsigned int*)lK[cur];
    const unsigned int* lV32 = (const unsigned int*)lV[cur];

    // ---- scores: S[16q x 32k] = (Q/8) K^T, two 16x16 tiles ----
    v8f st[2];
#pragma unroll
    for (int t = 0; t < 2; ++t) {
      v8f s = {};
#pragma unroll
      for (int c = 0; c < 2; ++c) {
        v8u kf;
#pragma unroll
        for (int v = 0; v < 8; ++v)
          kf[v] = lK32[((16 * t + fm) * 64 + 32 * c + hi * 16 + 2 * v) >> 1];
        s = wmma_bf16(qa[c], kf, s);
      }
      float mext = (1.0f - mbase[kk + 16 * t + fm]) * -3.0e38f;
#pragma unroll
      for (int v = 0; v < 8; ++v) s[v] = s[v] * 0.125f + mext;
      st[t] = s;
    }

    // ---- online softmax (row stats via 16-lane shuffles) ----
    float nm[8], sc[8];
#pragma unroll
    for (int v = 0; v < 8; ++v) {
      float x = fmaxf(st[0][v], st[1][v]);
#pragma unroll
      for (int o = 1; o < 16; o <<= 1) x = fmaxf(x, __shfl_xor(x, o, 16));
      nm[v] = fmaxf(mrow[v], x);
      sc[v] = __expf(mrow[v] - nm[v]);
      mrow[v] = nm[v];
    }
#pragma unroll
    for (int t = 0; t < 2; ++t)
#pragma unroll
      for (int v = 0; v < 8; ++v) st[t][v] = __expf(st[t][v] - nm[v]);
#pragma unroll
    for (int v = 0; v < 8; ++v) {
      float rs = st[0][v] + st[1][v];
#pragma unroll
      for (int o = 1; o < 16; o <<= 1) rs += __shfl_xor(rs, o, 16);
      lrow[v] = lrow[v] * sc[v] + rs;
    }
#pragma unroll
    for (int t4 = 0; t4 < 4; ++t4)
#pragma unroll
      for (int v = 0; v < 8; ++v) accO[t4][v] *= sc[v];

    // ---- P: C-layout -> A-fragment layout via per-wave LDS ----
    unsigned short* lp16 = (unsigned short*)lP[wid];
#pragma unroll
    for (int t = 0; t < 2; ++t)
#pragma unroll
      for (int v = 0; v < 8; ++v)
        lp16[(hi * 8 + v) * 32 + 16 * t + fm] = f2bfbits(st[t][v]);
    asm volatile("s_wait_dscnt 0" ::: "memory");
    v8u pa;
#pragma unroll
    for (int v = 0; v < 8; ++v) {
      int kd = hi * 16 + 2 * v;
      pa[v] = lP[wid][(fm * 32 + kd) >> 1];
    }

    // ---- O += P @ V from shared LDS tile ----
#pragma unroll
    for (int t4 = 0; t4 < 4; ++t4) {
      v8u vf;
#pragma unroll
      for (int v = 0; v < 8; ++v)
        vf[v] = lV32[((16 * t4 + fm) * 32 + hi * 16 + 2 * v) >> 1];
      accO[t4] = wmma_bf16(pa, vf, accO[t4]);
    }
  }

  // ---- normalize and store fp32 output ----
#pragma unroll
  for (int t4 = 0; t4 < 4; ++t4)
#pragma unroll
    for (int v = 0; v < 8; ++v) {
      int row = q0 + hi * 8 + v;
      int col = h * HD + 16 * t4 + fm;
      out[(size_t)(b * S + row) * H + col] = accO[t4][v] / lrow[v];
    }
}

// ---------------------------------------------------------------------------
extern "C" void kernel_launch(void* const* d_in, const int* in_sizes, int n_in,
                              void* d_out, int out_size, void* d_ws, size_t ws_size,
                              hipStream_t stream) {
  const float* hidden = (const float*)d_in[0];
  const float* mask   = (const float*)d_in[1];
  const float* Wq     = (const float*)d_in[2];
  const float* bq     = (const float*)d_in[3];
  const float* Wk     = (const float*)d_in[4];
  const float* bk     = (const float*)d_in[5];
  const float* Wv     = (const float*)d_in[6];
  const float* bv     = (const float*)d_in[7];
  float* out = (float*)d_out;

  const size_t elems = (size_t)4096 * 1024;      // B*S*H
  unsigned short* xbf = (unsigned short*)d_ws;                 // 4M
  unsigned short* wt  = xbf + elems;                           // 3M
  unsigned short* qws = wt + (size_t)3 * 1024 * 1024;          // 4M
  unsigned short* kws = qws + elems;                           // 4M
  unsigned short* vws = kws + elems;                           // 4M (transposed)

  prep_x<<<2048, 256, 0, stream>>>(hidden, xbf);
  prep_w<<<dim3(32, 32, 3), 256, 0, stream>>>(Wq, Wk, Wv, wt);
  qkv_gemm<<<dim3(32, 16, 3), 256, 0, stream>>>(
      xbf, wt, bq, bk, bv, qws, kws, vws);
  attn<<<dim3(16, 32), 256, 0, stream>>>(qws, kws, vws, mask, out);
}
